// SAModule_58600533786797
// MI455X (gfx1250) — compile-verified, matrix operand order checked by reading, not run on hardware
//
#include <hip/hip_runtime.h>
#include <hip/hip_bf16.h>

// ---------------- problem constants ----------------
#define B_     16
#define N_     4096
#define S_     1024
#define K_     32
#define C_IN_  64
#define C_OUT_ 128
#define FPAD_  96      // 67 (C_IN+3) padded to 3x32 for bf16 WMMA k-steps
#define H_     64

typedef __attribute__((ext_vector_type(16))) __bf16 v16bf;
typedef __attribute__((ext_vector_type(8)))  float  v8f;

// ---------------- WMMA helpers ----------------
__device__ inline v8f wmma_bf16(v16bf a, v16bf b, v8f c) {
  // D = A(16x32 bf16) x B(32x16 bf16) + C(16x16 f32)
  return __builtin_amdgcn_wmma_f32_16x16x32_bf16(
      /*neg_a=*/false, a, /*neg_b=*/false, b,
      /*c_mod=*/(short)0, c, /*reuse_a=*/false, /*reuse_b=*/false);
}

// Load a 16x32 bf16 fragment per the CDNA5 ISA 16-bit A-matrix layout:
//   lane L (mr=L&15, h=L>>4) holds row (rowbase+mr);
//   element e: K = kbase + (e/8)*16 + h*8 + (e%8)
// -> two contiguous 16-byte chunks at [row][kbase+h*8] and [row][kbase+16+h*8].
// Same mapping used for B fragments (lane -> column) against the transposed
// weight arrays W^T[n][k], so element e = W[K(e)][n].
__device__ inline v16bf load_frag(const __bf16* buf, int stride, int rowbase,
                                  int kbase, int lane) {
  const int h  = lane >> 4;
  const int mr = lane & 15;
  const __bf16* p = buf + (rowbase + mr) * stride + kbase + h * 8;
  v16bf a;
  uint4* ap = reinterpret_cast<uint4*>(&a);
  ap[0] = *reinterpret_cast<const uint4*>(p);
  ap[1] = *reinterpret_cast<const uint4*>(p + 16);
  return a;
}

__device__ inline v8f vzero8() {
  v8f z;
#pragma unroll
  for (int i = 0; i < 8; ++i) z[i] = 0.0f;
  return z;
}

// ---------------- kernel 0: weight prep (fp32 -> padded transposed bf16) ----
__global__ void prep_weights(const float* __restrict__ W1,
                             const float* __restrict__ W2,
                             const float* __restrict__ W3,
                             __bf16* __restrict__ w1t,   // [64][96]
                             __bf16* __restrict__ w2t,   // [64][64]
                             __bf16* __restrict__ w3t) { // [128][64]
  const int t = blockIdx.x * blockDim.x + threadIdx.x;
  const int stride = gridDim.x * blockDim.x;
  for (int i = t; i < H_ * FPAD_; i += stride) {
    int n = i / FPAD_, k = i % FPAD_;
    w1t[i] = (__bf16)((k < C_IN_ + 3) ? W1[k * H_ + n] : 0.0f);
  }
  for (int i = t; i < H_ * H_; i += stride) {
    int n = i / H_, k = i % H_;
    w2t[i] = (__bf16)W2[k * H_ + n];
  }
  for (int i = t; i < C_OUT_ * H_; i += stride) {
    int n = i / H_, k = i % H_;
    w3t[i] = (__bf16)W3[k * C_OUT_ + n];
  }
}

// ---------------- kernel 1: farthest point sampling -------------------------
// One 1024-thread workgroup per cloud. mind[] lives in LDS; block tree
// reduction keeps (max value, lowest index) to match jnp.argmax semantics.
__global__ __launch_bounds__(1024) void fps_kernel(
    const float* __restrict__ pos, float* __restrict__ qpos,
    int* __restrict__ nbatch) {
  __shared__ float mind[N_];
  __shared__ float rv[1024];
  __shared__ int   ri[1024];

  const int b = blockIdx.x;
  const int t = threadIdx.x;
  const float* P = pos + (size_t)b * N_ * 3;

  for (int i = t; i < N_; i += 1024) mind[i] = 3.0e38f;
  for (int s = t; s < S_; s += 1024) nbatch[b * S_ + s] = b;
  __syncthreads();

  int last = 0;
  for (int s = 0; s < S_; ++s) {
    const float lx = P[last * 3 + 0];
    const float ly = P[last * 3 + 1];
    const float lz = P[last * 3 + 2];
    if (t == 0) {
      float* qp = qpos + ((size_t)b * S_ + s) * 3;
      qp[0] = lx; qp[1] = ly; qp[2] = lz;
    }
    float bv = -1.0f; int bi = 0;
    for (int i = t; i < N_; i += 1024) {
      const float dx = P[i * 3 + 0] - lx;
      const float dy = P[i * 3 + 1] - ly;
      const float dz = P[i * 3 + 2] - lz;
      const float d = dx * dx + dy * dy + dz * dz;
      const float m = fminf(mind[i], d);
      mind[i] = m;
      if (m > bv) { bv = m; bi = i; }   // strict > keeps lowest index
    }
    rv[t] = bv; ri[t] = bi;
    __syncthreads();
#pragma unroll
    for (int off = 512; off > 0; off >>= 1) {
      if (t < off) {
        const float v2 = rv[t + off];
        const int   i2 = ri[t + off];
        if (v2 > rv[t] || (v2 == rv[t] && i2 < ri[t])) { rv[t] = v2; ri[t] = i2; }
      }
      __syncthreads();
    }
    last = ri[0];
    __syncthreads();
  }
}

// ---------------- kernel 2: radius ball query (first-K by index) ------------
// 8 waves/block, one query per wave; cloud staged in LDS; ordered append via
// wave32 ballot + prefix popcount. Mask packed as one uint (K==32).
__global__ __launch_bounds__(256) void ball_kernel(
    const float* __restrict__ pos, const float* __restrict__ qpos,
    int* __restrict__ nbr, unsigned* __restrict__ maskb) {
  __shared__ float sp[N_ * 3];
  const int t = threadIdx.x;
  const int qbase = blockIdx.x * 8;
  const int b = qbase / S_;
  const float* P = pos + (size_t)b * N_ * 3;
  for (int i = t; i < N_ * 3; i += 256) sp[i] = P[i];
  __syncthreads();

  const int wid = t >> 5, lane = t & 31;
  const int q = qbase + wid;
  const float qx = qpos[q * 3 + 0];
  const float qy = qpos[q * 3 + 1];
  const float qz = qpos[q * 3 + 2];
  const float r2 = 0.2f * 0.2f;

  int cnt = 0;
  for (int base = 0; base < N_ && cnt < K_; base += 32) {
    const int i = base + lane;
    const float dx = sp[i * 3 + 0] - qx;
    const float dy = sp[i * 3 + 1] - qy;
    const float dz = sp[i * 3 + 2] - qz;
    const bool valid = (dx * dx + dy * dy + dz * dz) <= r2;
    const unsigned m = (unsigned)__ballot(valid);
    const int slot = cnt + (int)__popc(m & ((1u << lane) - 1u));
    if (valid && slot < K_) nbr[(size_t)q * K_ + slot] = i;
    cnt += __popc(m);
  }
  if (cnt > K_) cnt = K_;
  for (int slot = lane; slot < K_; slot += 32)
    if (slot >= cnt) nbr[(size_t)q * K_ + slot] = 0;   // safe gather index
  if (lane == 0)
    maskb[q] = (cnt >= 32) ? 0xffffffffu : ((1u << cnt) - 1u);
}

// ---------------- kernel 3: fused gather + 3-layer MLP (WMMA) + masked max --
// 4 queries (128 rows) per block, 4 waves; wave w owns query w's 32 rows, so
// every A fragment a wave reads was written by itself (barriers only for
// safety). All intermediates stay in LDS/registers — nothing spills to HBM.
__global__ __launch_bounds__(128) void mlp_kernel(
    const float* __restrict__ x, const float* __restrict__ pos,
    const int* __restrict__ nbr, const unsigned* __restrict__ maskb,
    const float* __restrict__ qpos,
    const __bf16* __restrict__ w1t, const __bf16* __restrict__ w2t,
    const __bf16* __restrict__ w3t,
    const float* __restrict__ b1, const float* __restrict__ b2,
    const float* __restrict__ b3, float* __restrict__ out) {
  __shared__ __bf16 sA[128 * FPAD_];   // input features  [row][96]
  __shared__ __bf16 sH1[128 * H_];     // layer-1 output  [row][64]
  __shared__ __bf16 sH2[128 * H_];     // layer-2 output  [row][64]

  const int t = threadIdx.x;
  const int lane = t & 31;
  const int wid = t >> 5;
  const int q0 = blockIdx.x * 4;

  // ---- stage 0: gather x_j and relative positions, convert to bf16 ----
  {
    const int r = t;                       // one row per thread
    const int q = q0 + (r >> 5);
    const int k = r & 31;
    const int b = q / S_;
    const int nb = nbr[(size_t)q * K_ + k];
    const size_t gi = (size_t)b * N_ + (size_t)nb;
    const float4* xr = reinterpret_cast<const float4*>(x + gi * C_IN_);
    __bf16* dst = &sA[r * FPAD_];
#pragma unroll
    for (int j = 0; j < 16; ++j) {
      const float4 v = xr[j];
      dst[4 * j + 0] = (__bf16)v.x;
      dst[4 * j + 1] = (__bf16)v.y;
      dst[4 * j + 2] = (__bf16)v.z;
      dst[4 * j + 3] = (__bf16)v.w;
    }
    const float qx = qpos[q * 3 + 0], qy = qpos[q * 3 + 1], qz = qpos[q * 3 + 2];
    dst[64] = (__bf16)(pos[gi * 3 + 0] - qx);
    dst[65] = (__bf16)(pos[gi * 3 + 1] - qy);
    dst[66] = (__bf16)(pos[gi * 3 + 2] - qz);
#pragma unroll
    for (int j = 67; j < FPAD_; ++j) dst[j] = (__bf16)0.0f;
  }
  __syncthreads();

  const int q = q0 + wid;
  const int m0 = wid * 2;           // this wave's first 16-row tile (of 8)
  const int h = lane >> 4;
  const int cl = lane & 15;

  // ---- layer 1: [32x96] x [96x64], relu, -> sH1 ----
  for (int n = 0; n < 4; ++n) {
    const float bias = b1[n * 16 + cl];
#pragma unroll
    for (int mi = 0; mi < 2; ++mi) {
      v8f acc = vzero8();
#pragma unroll
      for (int kk = 0; kk < 3; ++kk) {
        const v16bf a  = load_frag(sA,  FPAD_, (m0 + mi) * 16, kk * 32, lane);
        const v16bf bf = load_frag(w1t, FPAD_, n * 16,         kk * 32, lane);
        acc = wmma_bf16(a, bf, acc);
      }
#pragma unroll
      for (int v = 0; v < 8; ++v) {
        const int row = (m0 + mi) * 16 + v + 8 * h;
        sH1[row * H_ + n * 16 + cl] = (__bf16)fmaxf(acc[v] + bias, 0.0f);
      }
    }
  }
  __syncthreads();

  // ---- layer 2: [32x64] x [64x64], relu, -> sH2 ----
  for (int n = 0; n < 4; ++n) {
    const float bias = b2[n * 16 + cl];
#pragma unroll
    for (int mi = 0; mi < 2; ++mi) {
      v8f acc = vzero8();
#pragma unroll
      for (int kk = 0; kk < 2; ++kk) {
        const v16bf a  = load_frag(sH1, H_, (m0 + mi) * 16, kk * 32, lane);
        const v16bf bf = load_frag(w2t, H_, n * 16,         kk * 32, lane);
        acc = wmma_bf16(a, bf, acc);
      }
#pragma unroll
      for (int v = 0; v < 8; ++v) {
        const int row = (m0 + mi) * 16 + v + 8 * h;
        sH2[row * H_ + n * 16 + cl] = (__bf16)fmaxf(acc[v] + bias, 0.0f);
      }
    }
  }
  __syncthreads();

  // ---- layer 3: [32x64] x [64x128] + fused masked max over K, relu ----
  const unsigned mb = maskb[q];
  for (int n = 0; n < 8; ++n) {
    v8f acc0 = vzero8();   // rows k = 0..15  (tile m0)
    v8f acc1 = vzero8();   // rows k = 16..31 (tile m0+1)
#pragma unroll
    for (int kk = 0; kk < 2; ++kk) {
      const v16bf bf = load_frag(w3t, H_, n * 16, kk * 32, lane);
      const v16bf a0 = load_frag(sH2, H_, m0 * 16,       kk * 32, lane);
      acc0 = wmma_bf16(a0, bf, acc0);
      const v16bf a1 = load_frag(sH2, H_, (m0 + 1) * 16, kk * 32, lane);
      acc1 = wmma_bf16(a1, bf, acc1);
    }
    // In-register masked max: lane holds rows {h*8..h*8+7} of each half-tile.
    float best = -3.4e38f;
#pragma unroll
    for (int v = 0; v < 8; ++v) {
      const int k0 = v + 8 * h;        // k in 0..15
      const int k1 = 16 + v + 8 * h;   // k in 16..31
      if ((mb >> k0) & 1u) best = fmaxf(best, acc0[v]);
      if ((mb >> k1) & 1u) best = fmaxf(best, acc1[v]);
    }
    best = fmaxf(best, __shfl_xor(best, 16, 32));   // merge the two row halves
    const int col = n * 16 + cl;
    const float res = (mb == 0u) ? 0.0f : fmaxf(best + b3[col], 0.0f);
    if (lane < 16) out[(size_t)q * C_OUT_ + col] = res;
  }
}

// ---------------- launch ----------------------------------------------------
extern "C" void kernel_launch(void* const* d_in, const int* in_sizes, int n_in,
                              void* d_out, int out_size, void* d_ws,
                              size_t ws_size, hipStream_t stream) {
  const float* x   = (const float*)d_in[0];
  const float* pos = (const float*)d_in[1];
  // d_in[2] = batch (implied by equal-size clouds; unused)
  const float* W1 = (const float*)d_in[3];
  const float* b1 = (const float*)d_in[4];
  const float* W2 = (const float*)d_in[5];
  const float* b2 = (const float*)d_in[6];
  const float* W3 = (const float*)d_in[7];
  const float* b3 = (const float*)d_in[8];

  // d_out = [out (B*S*128 f32)] [qpos (B*S*3 f32)] [new_batch (B*S i32 bits)]
  float* out   = (float*)d_out;
  float* qpos  = out + (size_t)B_ * S_ * C_OUT_;
  int*   nbat  = (int*)(qpos + (size_t)B_ * S_ * 3);

  // workspace: nbr ints, mask words, bf16 transposed weights
  char* ws = (char*)d_ws;
  int*      nbr   = (int*)ws;                                    // 2 MiB
  unsigned* maskb = (unsigned*)(ws + (size_t)B_ * S_ * K_ * 4);  // 64 KiB
  __bf16*   w1t   = (__bf16*)((char*)maskb + (size_t)B_ * S_ * 4);
  __bf16*   w2t   = w1t + H_ * FPAD_;
  __bf16*   w3t   = w2t + H_ * H_;

  prep_weights<<<16, 256, 0, stream>>>(W1, W2, W3, w1t, w2t, w3t);
  fps_kernel<<<B_, 1024, 0, stream>>>(pos, qpos, nbat);
  ball_kernel<<<(B_ * S_) / 8, 256, 0, stream>>>(pos, qpos, nbr, maskb);
  mlp_kernel<<<(B_ * S_) / 4, 128, 0, stream>>>(x, pos, nbr, maskb, qpos,
                                                w1t, w2t, w3t, b1, b2, b3, out);
}
